// DefuzzyLayer_23708219474208
// MI455X (gfx1250) — matrix operand order repaired and checked by reading, not compiled, not run on hardware
//
#include <hip/hip_runtime.h>
#include <stdint.h>

// ---- problem dims (from reference) ----
#define M_DIM 8192
#define K_DIM 4096
#define N_DIM 4096

// ---- tiling ----
#define BM 128
#define BN 128
#define BK 32
#define LDA 40   // LDS row stride in bf16 elems (32 + 8 pad -> conflict-free b128 reads)
#define LDB 40
#define NT (K_DIM / BK)   // 128 K-steps

typedef __attribute__((ext_vector_type(16))) __bf16 v16bf;
typedef __attribute__((ext_vector_type(8)))  __bf16 v8bf;
typedef __attribute__((ext_vector_type(4)))  __bf16 v4bf;
typedef __attribute__((ext_vector_type(8)))  float  v8f;

__global__ __launch_bounds__(256) void defuzzy_wmma_bf16(
    const float* __restrict__ A,   // [M_DIM, K_DIM] row-major
    const float* __restrict__ B,   // [K_DIM, N_DIM] row-major
    float* __restrict__ C)         // [M_DIM, N_DIM] row-major
{
  // double-buffered tiles: A row-major (M x K), B transposed (N x K)
  __shared__ __align__(16) __bf16 smA[2 * BM * LDA];
  __shared__ __align__(16) __bf16 smB[2 * BN * LDB];

  const int tid  = threadIdx.x;
  const int lane = tid & 31;
  const int wave = tid >> 5;
  const int wm   = wave & 3;   // 4 wave-rows * 32 rows = 128
  const int wn   = wave >> 2;  // 2 wave-cols * 64 cols = 128

  const int m0 = blockIdx.y * BM;
  const int n0 = blockIdx.x * BN;

  // staging maps
  const int a_c  = tid & 7;    // which float4 in the 32-wide K slice
  const int a_r  = tid >> 3;   // base row 0..31 (covers +32,+64,+96)
  const int b_n  = tid & 127;  // output column within tile
  const int b_kq = tid >> 7;   // K half: 0 -> K0..15, 1 -> K16..31

  const int half = (lane < 16) ? 0 : 1;
  const int l15  = lane & 15;

  // prefetch registers (next tile, raw fp32)
  float4 apf[4];
  float  bpf[16];

  auto load_tiles = [&](int k0) {
#pragma unroll
    for (int i = 0; i < 4; ++i)
      apf[i] = *(const float4*)(A + (size_t)(m0 + a_r + 32 * i) * K_DIM +
                                k0 + a_c * 4);
#pragma unroll
    for (int kk = 0; kk < 16; ++kk)
      bpf[kk] = B[(size_t)(k0 + b_kq * 16 + kk) * N_DIM + n0 + b_n];
  };

  // fp32 -> bf16 with native RNE convert (fptrunc -> v_cvt_*bf16*), then
  // wide LDS stores. No manual bit twiddling: avoids the b16 half-move storm.
  auto store_tiles = [&](int buf) {
    __bf16* sa = smA + buf * (BM * LDA);
    __bf16* sb = smB + buf * (BN * LDB);
#pragma unroll
    for (int i = 0; i < 4; ++i) {
      v4bf q;
      q.x = (__bf16)apf[i].x;
      q.y = (__bf16)apf[i].y;
      q.z = (__bf16)apf[i].z;
      q.w = (__bf16)apf[i].w;
      *(v4bf*)&sa[(a_r + 32 * i) * LDA + a_c * 4] = q;   // 8B store
    }
    v8bf q0, q1;
#pragma unroll
    for (int i = 0; i < 8; ++i) q0[i] = (__bf16)bpf[i];
#pragma unroll
    for (int i = 0; i < 8; ++i) q1[i] = (__bf16)bpf[8 + i];
    *(v8bf*)&sb[b_n * LDB + b_kq * 16]     = q0;          // 16B store
    *(v8bf*)&sb[b_n * LDB + b_kq * 16 + 8] = q1;          // 16B store
  };

  v8f acc[2][4];
#pragma unroll
  for (int i = 0; i < 2; ++i)
#pragma unroll
    for (int j = 0; j < 4; ++j) {
      v8f z = {0.f, 0.f, 0.f, 0.f, 0.f, 0.f, 0.f, 0.f};
      acc[i][j] = z;
    }

  // ---- prologue: stage K-step 0 into buffer 0 ----
  load_tiles(0);
  store_tiles(0);
  __syncthreads();

#pragma unroll 1
  for (int kt = 0; kt < NT; ++kt) {
    const int cur = kt & 1;
    const bool has_next = (kt + 1) < NT;

    // issue next tile's global loads FIRST so HBM latency hides under WMMA
    if (has_next) load_tiles((kt + 1) * BK);

    // ---- load WMMA fragments from LDS buffer `cur` ----
    const __bf16* sa = smA + cur * (BM * LDA);
    const __bf16* sb = smB + cur * (BN * LDB);

    // A 16x32 bf16 layout: lanes 0-15 row M=l, elems = K{0..7,16..23};
    //                      lanes 16-31 row M=l-16, elems = K{8..15,24..31}
    v16bf afr[2];
#pragma unroll
    for (int tm = 0; tm < 2; ++tm) {
      const int arow = wm * 32 + tm * 16 + l15;
      const __bf16* p = &sa[arow * LDA + half * 8];
      ((v8bf*)&afr[tm])[0] = *(const v8bf*)p;
      ((v8bf*)&afr[tm])[1] = *(const v8bf*)(p + 16);
    }
    // B 32x16 bf16 layout: lanes 0-15 col N=l, elems = K{0..7,16..23};
    //                      lanes 16-31 col N=l-16, elems = K{8..15,24..31}
    v16bf bfr[4];
#pragma unroll
    for (int tn = 0; tn < 4; ++tn) {
      const int bcol = wn * 64 + tn * 16 + l15;
      const __bf16* p = &sb[bcol * LDB + half * 8];
      ((v8bf*)&bfr[tn])[0] = *(const v8bf*)p;
      ((v8bf*)&bfr[tn])[1] = *(const v8bf*)(p + 16);
    }

    // ---- 8 WMMAs: D = A x B + C, fp32 accumulate ----
#pragma unroll
    for (int tm = 0; tm < 2; ++tm)
#pragma unroll
      for (int tn = 0; tn < 4; ++tn)
        acc[tm][tn] = __builtin_amdgcn_wmma_f32_16x16x32_bf16(
            /*neg_a=*/false, afr[tm], /*neg_b=*/false, bfr[tn],
            /*c_mod=*/(short)0, acc[tm][tn],
            /*reuse_a=*/false, /*reuse_b=*/false);

    // ---- convert + store next tile into the other buffer ----
    if (has_next) store_tiles(cur ^ 1);

    __syncthreads();
  }

  // ---- epilogue: C/D layout: VGPR r -> row tile_m + r + (lane<16 ? 0 : 8),
  //                            lane&15 -> column ----
#pragma unroll
  for (int tm = 0; tm < 2; ++tm) {
#pragma unroll
    for (int tn = 0; tn < 4; ++tn) {
      const int col     = n0 + wn * 64 + tn * 16 + l15;
      const int rowbase = m0 + wm * 32 + tm * 16 + half * 8;
#pragma unroll
      for (int r = 0; r < 8; ++r) {
        C[(size_t)(rowbase + r) * N_DIM + col] = acc[tm][tn][r];
      }
    }
  }
}

extern "C" void kernel_launch(void* const* d_in, const int* in_sizes, int n_in,
                              void* d_out, int out_size, void* d_ws, size_t ws_size,
                              hipStream_t stream) {
  const float* x = (const float*)d_in[0];             // [8192, 4096]
  const float* w = (const float*)d_in[1];             // [4096, 4096]
  float* out = (float*)d_out;                         // [8192, 4096]
  (void)in_sizes; (void)n_in; (void)out_size; (void)d_ws; (void)ws_size;

  dim3 grid(N_DIM / BN, M_DIM / BM);  // 32 x 64 = 2048 blocks
  defuzzy_wmma_bf16<<<grid, 256, 0, stream>>>(x, w, out);
}
